// _Polynormer_15367392985095
// MI455X (gfx1250) — matrix operand order repaired and checked by reading, not compiled, not run on hardware
//
#include <hip/hip_runtime.h>
#include <hip/hip_bf16.h>
#include <stdint.h>

typedef __attribute__((ext_vector_type(16))) __bf16 v16bf;
typedef __attribute__((ext_vector_type(8)))  float  v8f;

#define Nn   8192
#define Ee   131072
#define INC  128
#define HID  256
#define HEADS 4
#define DH   64
#define OUTC 64

__device__ __forceinline__ unsigned short f2bf(float f) {
  union { float f; unsigned u; } x; x.f = f;
  unsigned u = x.u;
  unsigned r = u + 0x7FFFu + ((u >> 16) & 1u);
  return (unsigned short)(r >> 16);
}

union Frag { v16bf v; uint4 q[2]; unsigned short u[16]; };

__device__ __forceinline__ v8f wmma_bf16(v16bf a, v16bf b, v8f c) {
  return __builtin_amdgcn_wmma_f32_16x16x32_bf16(false, a, false, b, (short)0, c, false, false);
}

// ---------------- weight pack: f32 [K,N] row-major -> bf16 Bp[(k/16)][n][16] ----------------
__global__ void k_pack(const float* __restrict__ W, unsigned short* __restrict__ Bp, int K, int N) {
  int idx = blockIdx.x * 256 + threadIdx.x;
  if (idx >= K * N) return;
  int k = idx / N, n = idx % N;
  Bp[((size_t)(k >> 4) * N + n) * 16 + (k & 15)] = f2bf(W[idx]);
}

__global__ void k_cvt(const float* __restrict__ s, unsigned short* __restrict__ d, int count) {
  int i = blockIdx.x * 256 + threadIdx.x;
  if (i < count) d[i] = f2bf(s[i]);
}

// ---------------- degree / norm ----------------
__global__ void k_deginit(float* deg) {
  int i = blockIdx.x * 256 + threadIdx.x;
  if (i < Nn) deg[i] = 1.0f;   // self loop counted up-front
}
__global__ void k_degacc(const int* __restrict__ row, float* deg) {
  int e = blockIdx.x * 256 + threadIdx.x;
  if (e < Ee) atomicAdd(&deg[row[e]], 1.0f);
}
__global__ void k_norm(float* deg) {
  int i = blockIdx.x * 256 + threadIdx.x;
  if (i < Nn) deg[i] = rsqrtf(deg[i]);   // deg >= 1 always
}

// ---------------- bf16 WMMA GEMM: one wave per 16(M) x 64(N) output tile ----------------
// K templated + fully unrolled: every fragment is a fresh SSA value, so the
// scheduler can prefetch iteration i+1 loads above iteration i WMMAs instead
// of serializing on a reused register block.
// mode 0: outF = A@Bp (+bias)   fp32 row-major
// mode 1: outB = bf16(A@Bp)     row-major
// mode 2: outB = bf16(A@Bp)^T   (column-major store, stride M)
template<int KT>
__global__ __launch_bounds__(32) void k_gemm(const unsigned short* __restrict__ A,
    const unsigned short* __restrict__ Bp, const float* __restrict__ bias,
    float* __restrict__ outF, unsigned short* __restrict__ outB,
    int M, int N, int mode)
{
  const int lane = threadIdx.x & 31;
  const int lo = lane & 15, hi = lane >> 4;
  const int m = blockIdx.y * 16 + lo;
  const int nBase = blockIdx.x * 64;
  v8f c0 = {}, c1 = {}, c2 = {}, c3 = {};
  const unsigned short* pa0 = A + (size_t)m * KT + 16 * hi;
  const unsigned short* pb0 = Bp + ((size_t)hi * N + nBase + lo) * 16;
#pragma unroll
  for (int kk = 0; kk < KT; kk += 32) {
    Frag a;
    const unsigned short* pa = pa0 + kk;
    a.q[0] = *(const uint4*)pa;          // j=0..7  -> k = kk+16*hi+j
    a.q[1] = *(const uint4*)(pa + 8);    // j=8..15 -> k = kk+16*hi+j
    const unsigned short* pb = pb0 + (size_t)(kk >> 4) * N * 16;
    Frag b0, b1, b2, b3;
    b0.q[0] = ((const uint4*)pb)[0];         b0.q[1] = ((const uint4*)pb)[1];
    b1.q[0] = ((const uint4*)(pb + 256))[0]; b1.q[1] = ((const uint4*)(pb + 256))[1];
    b2.q[0] = ((const uint4*)(pb + 512))[0]; b2.q[1] = ((const uint4*)(pb + 512))[1];
    b3.q[0] = ((const uint4*)(pb + 768))[0]; b3.q[1] = ((const uint4*)(pb + 768))[1];
    c0 = wmma_bf16(a.v, b0.v, c0);
    c1 = wmma_bf16(a.v, b1.v, c1);
    c2 = wmma_bf16(a.v, b2.v, c2);
    c3 = wmma_bf16(a.v, b3.v, c3);
  }
  const int rowBase = blockIdx.y * 16 + 8 * hi;
  v8f* accs[4] = { &c0, &c1, &c2, &c3 };
#pragma unroll
  for (int t = 0; t < 4; ++t) {
    const int n = nBase + t * 16 + lo;
    v8f c = *accs[t];
    if (mode == 0) {
      float bb = bias ? bias[n] : 0.f;
#pragma unroll
      for (int r = 0; r < 8; ++r)
        outF[(size_t)(rowBase + r) * N + n] = c[r] + bb;
    } else if (mode == 1) {
#pragma unroll
      for (int r = 0; r < 8; ++r)
        outB[(size_t)(rowBase + r) * N + n] = f2bf(c[r]);
    } else {
#pragma unroll
      for (int r = 0; r < 8; ++r)
        outB[(size_t)n * M + rowBase + r] = f2bf(c[r]);
    }
  }
}

// ---------------- local propagation: self-loop init + edge scatter ----------------
__global__ void k_selfagg(const float* __restrict__ z, const float* __restrict__ norm,
                          float* __restrict__ agg) {
  int i = blockIdx.x * 256 + threadIdx.x;
  if (i < Nn * HID) { float nv = norm[i >> 8]; agg[i] = nv * nv * z[i]; }
}
__global__ __launch_bounds__(HID) void k_edgeagg(const float* __restrict__ z,
    const float* __restrict__ norm, const int* __restrict__ row,
    const int* __restrict__ col, float* __restrict__ agg) {
  int e = blockIdx.x;
  int c = threadIdx.x;
  int rj = row[e], ci = col[e];
  float w = norm[rj] * norm[ci];
  atomicAdd(&agg[(size_t)ci * HID + c], w * z[(size_t)rj * HID + c]);
}

// ---------------- LayerNorm(a + res) -> h (fp32) + hbf (bf16) ----------------
__global__ __launch_bounds__(HID) void k_ln(const float* __restrict__ a, const float* __restrict__ res,
    const float* __restrict__ g, const float* __restrict__ b, int goff,
    float* __restrict__ h, unsigned short* __restrict__ hbf)
{
  __shared__ float red[HID];
  int row = blockIdx.x, c = threadIdx.x;
  size_t idx = (size_t)row * HID + c;
  float v = a[idx] + res[idx];
  red[c] = v; __syncthreads();
  for (int s = HID / 2; s > 0; s >>= 1) { if (c < s) red[c] += red[c + s]; __syncthreads(); }
  float mean = red[0] * (1.f / HID);
  __syncthreads();
  float d = v - mean;
  red[c] = d * d; __syncthreads();
  for (int s = HID / 2; s > 0; s >>= 1) { if (c < s) red[c] += red[c + s]; __syncthreads(); }
  float var = red[0] * (1.f / HID);
  float o = d * rsqrtf(var + 1e-5f) * g[goff + c] + b[goff + c];
  h[idx] = o;
  hbf[idx] = f2bf(o);
}

// ---------------- flash attention ----------------
// Block: 8 waves (256 threads) = 128 query rows of one head.
// K-tile (32x64) and Vt-tile (64x32) staged cooperatively into LDS, ping-pong
// double-buffered: one barrier per step, next tile's global loads overlap the
// current tile's WMMA + softmax compute. 8x L2 traffic reduction vs per-wave.
__global__ __launch_bounds__(256) void k_attn(const unsigned short* __restrict__ Q,
    const unsigned short* __restrict__ Kb, const unsigned short* __restrict__ Vt,
    const float* __restrict__ beta, int layer, unsigned short* __restrict__ O)
{
  __shared__ __attribute__((aligned(16))) unsigned short ldsK[2][32 * 64];   // [buf][key][d]
  __shared__ __attribute__((aligned(16))) unsigned short ldsV[2][64 * 32];   // [buf][d][key]
  __shared__ __attribute__((aligned(16))) unsigned short ldsP[8 * 16 * 32];  // per-wave P tile

  const int head = blockIdx.y;
  const int tid  = threadIdx.x;
  const int wave = tid >> 5;
  const int lane = tid & 31;
  const int lo = lane & 15, hi = lane >> 4;
  const int qBase = blockIdx.x * 128 + wave * 16;
  unsigned short* myP = ldsP + wave * (16 * 32);

  float bt = beta[layer];
  float bscale = (bt < 0.f ? log1pf(__expf(bt)) : bt) * 0.125f;  // softplus(beta)/sqrt(DH)

  // Q in A-fragment layout, K-dim = DH=64 -> two fragments
  Frag q0, q1;
  { const unsigned short* qr = Q + (size_t)(qBase + lo) * (HEADS * DH) + head * DH;
    q0.q[0] = *(const uint4*)(qr + 16 * hi);
    q0.q[1] = *(const uint4*)(qr + 8 + 16 * hi);
    q1.q[0] = *(const uint4*)(qr + 32 + 16 * hi);
    q1.q[1] = *(const uint4*)(qr + 40 + 16 * hi); }

  // staging indices (one uint4 per thread per tile)
  const int skey = tid >> 3, sq8 = tid & 7;                       // K tile
  const int sd   = tid >> 2, sq4 = tid & 3;                       // Vt tile
  const unsigned short* kSrcBase = Kb + (size_t)skey * (HEADS * DH) + head * DH + sq8 * 8;
  const unsigned short* vSrcBase = Vt + (size_t)(head * DH + sd) * Nn + sq4 * 8;
  const int kDstOff = skey * 64 + sq8 * 8;
  const int vDstOff = sd * 32 + sq4 * 8;

  float mrow[8], lrow[8];
  v8f acc0 = {}, acc1 = {}, acc2 = {}, acc3 = {};
#pragma unroll
  for (int r = 0; r < 8; ++r) { mrow[r] = -1e30f; lrow[r] = 0.f; }

  // prologue: stage first tile into buffer 0
  *(uint4*)&ldsK[0][kDstOff] = *(const uint4*)kSrcBase;
  *(uint4*)&ldsV[0][vDstOff] = *(const uint4*)vSrcBase;

  for (int kb = 0; kb < Nn; kb += 32) {
    const int cur = (kb >> 5) & 1;
    __syncthreads();   // staging of `cur` complete; compute on cur-1 finished everywhere

    // prefetch next tile into the other buffer (overlaps with compute below)
    if (kb + 32 < Nn) {
      *(uint4*)&ldsK[cur ^ 1][kDstOff] = *(const uint4*)(kSrcBase + (size_t)(kb + 32) * (HEADS * DH));
      *(uint4*)&ldsV[cur ^ 1][vDstOff] = *(const uint4*)(vSrcBase + kb + 32);
    }

    // S = Q K^T for 32 keys (two 16-col tiles), K-dim 64
    v8f s0 = {}, s1 = {};
    { const unsigned short* kr = &ldsK[cur][(lo) * 64 + 16 * hi];
      Frag b0, b1;
      b0.q[0] = *(const uint4*)kr;        b0.q[1] = *(const uint4*)(kr + 8);
      b1.q[0] = *(const uint4*)(kr + 32); b1.q[1] = *(const uint4*)(kr + 40);
      s0 = wmma_bf16(q0.v, b0.v, s0);
      s0 = wmma_bf16(q1.v, b1.v, s0); }
    { const unsigned short* kr = &ldsK[cur][(16 + lo) * 64 + 16 * hi];
      Frag b0, b1;
      b0.q[0] = *(const uint4*)kr;        b0.q[1] = *(const uint4*)(kr + 8);
      b1.q[0] = *(const uint4*)(kr + 32); b1.q[1] = *(const uint4*)(kr + 40);
      s1 = wmma_bf16(q0.v, b0.v, s1);
      s1 = wmma_bf16(q1.v, b1.v, s1); }

    // online softmax (row m = r + 8*hi, cols across lanes 0..15 of group)
#pragma unroll
    for (int r = 0; r < 8; ++r) {
      float a0 = s0[r] * bscale, a1 = s1[r] * bscale;
      float rmax = fmaxf(a0, a1);
      for (int msk = 1; msk < 16; msk <<= 1) rmax = fmaxf(rmax, __shfl_xor(rmax, msk, 32));
      float nm = fmaxf(mrow[r], rmax);
      float corr = __expf(mrow[r] - nm);
      float p0 = __expf(a0 - nm);
      float p1 = __expf(a1 - nm);
      float rs = p0 + p1;
      for (int msk = 1; msk < 16; msk <<= 1) rs += __shfl_xor(rs, msk, 32);
      lrow[r] = lrow[r] * corr + rs;
      mrow[r] = nm;
      acc0[r] *= corr; acc1[r] *= corr; acc2[r] *= corr; acc3[r] *= corr;
      myP[(r + 8 * hi) * 32 + lo]      = f2bf(p0);   // P tile, row-major (wave-private)
      myP[(r + 8 * hi) * 32 + 16 + lo] = f2bf(p1);
    }

    // reload P in A-fragment layout (same wave: DS ops are in-order)
    Frag p;
    p.q[0] = *(const uint4*)&myP[lo * 32 + 16 * hi];
    p.q[1] = *(const uint4*)&myP[lo * 32 + 8 + 16 * hi];

#pragma unroll
    for (int dg = 0; dg < 4; ++dg) {
      const unsigned short* vr = &ldsV[cur][(dg * 16 + lo) * 32 + 16 * hi];
      Frag bv; bv.q[0] = *(const uint4*)vr; bv.q[1] = *(const uint4*)(vr + 8);
      v8f* ap = (dg == 0) ? &acc0 : (dg == 1) ? &acc1 : (dg == 2) ? &acc2 : &acc3;
      *ap = wmma_bf16(p.v, bv.v, *ap);
    }
  }

#pragma unroll
  for (int r = 0; r < 8; ++r) {
    float inv = 1.f / lrow[r];
    size_t orow = (size_t)(qBase + r + 8 * hi) * (HEADS * DH) + head * DH + lo;
    O[orow]      = f2bf(acc0[r] * inv);
    O[orow + 16] = f2bf(acc1[r] * inv);
    O[orow + 32] = f2bf(acc2[r] * inv);
    O[orow + 48] = f2bf(acc3[r] * inv);
  }
}

extern "C" void kernel_launch(void* const* d_in, const int* in_sizes, int n_in,
                              void* d_out, int out_size, void* d_ws, size_t ws_size,
                              hipStream_t stream)
{
  (void)in_sizes; (void)n_in; (void)out_size; (void)ws_size;
  const float* x      = (const float*)d_in[0];
  const int*   eidx   = (const int*)d_in[1];
  const float* lin_w  = (const float*)d_in[2];
  const float* lin_b  = (const float*)d_in[3];
  const float* loc_w  = (const float*)d_in[4];
  const float* loc_g  = (const float*)d_in[5];
  const float* loc_bb = (const float*)d_in[6];
  const float* wq     = (const float*)d_in[7];
  const float* wk     = (const float*)d_in[8];
  const float* wv     = (const float*)d_in[9];
  const float* wf     = (const float*)d_in[10];
  const float* bf_    = (const float*)d_in[11];
  const float* beta   = (const float*)d_in[12];
  const float* glob_g = (const float*)d_in[13];
  const float* glob_b = (const float*)d_in[14];
  const float* out_w  = (const float*)d_in[15];
  const float* out_b  = (const float*)d_in[16];

  char* ws = (char*)d_ws;
  size_t off = 0;
  auto carve = [&](size_t bytes) -> void* {
    void* p = ws + off; off += (bytes + 255) & ~(size_t)255; return p;
  };
  float* norm = (float*)carve((size_t)Nn * 4);
  float* h    = (float*)carve((size_t)Nn * HID * 4);
  float* tmp  = (float*)carve((size_t)Nn * HID * 4);
  unsigned short* hbf  = (unsigned short*)carve((size_t)Nn * HID * 2);
  unsigned short* qbf  = (unsigned short*)carve((size_t)Nn * HID * 2);
  unsigned short* kbf  = (unsigned short*)carve((size_t)Nn * HID * 2);
  unsigned short* vtbf = (unsigned short*)carve((size_t)Nn * HID * 2);
  unsigned short* obf  = (unsigned short*)carve((size_t)Nn * HID * 2);
  float* agg = (float*)qbf;              // alias: local phase only (spans qbf+kbf = 8MB)
  unsigned short* xbf = obf;             // alias: used before attention phase
  unsigned short* linp = (unsigned short*)carve((size_t)INC * HID * 2);
  unsigned short* locp = (unsigned short*)carve((size_t)2 * HID * HID * 2);
  unsigned short* wqp  = (unsigned short*)carve((size_t)2 * HID * HID * 2);
  unsigned short* wkp  = (unsigned short*)carve((size_t)2 * HID * HID * 2);
  unsigned short* wvp  = (unsigned short*)carve((size_t)2 * HID * HID * 2);
  unsigned short* wfp  = (unsigned short*)carve((size_t)2 * HID * HID * 2);
  unsigned short* outp = (unsigned short*)carve((size_t)HID * OUTC * 2);

  // weight packing (bf16, B-layout)
  k_pack<<<(INC * HID + 255) / 256, 256, 0, stream>>>(lin_w, linp, INC, HID);
  for (int i = 0; i < 2; ++i) {
    size_t o = (size_t)i * HID * HID;
    k_pack<<<(HID * HID + 255) / 256, 256, 0, stream>>>(loc_w + o, locp + o, HID, HID);
    k_pack<<<(HID * HID + 255) / 256, 256, 0, stream>>>(wq + o, wqp + o, HID, HID);
    k_pack<<<(HID * HID + 255) / 256, 256, 0, stream>>>(wk + o, wkp + o, HID, HID);
    k_pack<<<(HID * HID + 255) / 256, 256, 0, stream>>>(wv + o, wvp + o, HID, HID);
    k_pack<<<(HID * HID + 255) / 256, 256, 0, stream>>>(wf + o, wfp + o, HID, HID);
  }
  k_pack<<<(HID * OUTC + 255) / 256, 256, 0, stream>>>(out_w, outp, HID, OUTC);

  // degrees -> norm (self loop pre-counted)
  k_deginit<<<(Nn + 255) / 256, 256, 0, stream>>>(norm);
  k_degacc<<<(Ee + 255) / 256, 256, 0, stream>>>(eidx, norm);
  k_norm<<<(Nn + 255) / 256, 256, 0, stream>>>(norm);

  // h = x @ Win + b
  k_cvt<<<(Nn * INC + 255) / 256, 256, 0, stream>>>(x, xbf, Nn * INC);
  k_gemm<INC><<<dim3(HID / 64, Nn / 16), 32, 0, stream>>>(xbf, linp, lin_b, h, nullptr, Nn, HID, 0);
  k_cvt<<<(Nn * HID + 255) / 256, 256, 0, stream>>>(h, hbf, Nn * HID);

  // local propagation layers
  for (int i = 0; i < 2; ++i) {
    k_gemm<HID><<<dim3(HID / 64, Nn / 16), 32, 0, stream>>>(hbf, locp + (size_t)i * HID * HID,
                                                            nullptr, tmp, nullptr, Nn, HID, 0);
    k_selfagg<<<(Nn * HID + 255) / 256, 256, 0, stream>>>(tmp, norm, agg);
    k_edgeagg<<<Ee, HID, 0, stream>>>(tmp, norm, eidx, eidx + Ee, agg);
    k_ln<<<Nn, HID, 0, stream>>>(agg, h, loc_g, loc_bb, i * HID, h, hbf);
  }

  // global attention layers
  for (int i = 0; i < 2; ++i) {
    size_t o = (size_t)i * HID * HID;
    k_gemm<HID><<<dim3(HID / 64, Nn / 16), 32, 0, stream>>>(hbf, wqp + o, nullptr, nullptr, qbf,  Nn, HID, 1);
    k_gemm<HID><<<dim3(HID / 64, Nn / 16), 32, 0, stream>>>(hbf, wkp + o, nullptr, nullptr, kbf,  Nn, HID, 1);
    k_gemm<HID><<<dim3(HID / 64, Nn / 16), 32, 0, stream>>>(hbf, wvp + o, nullptr, nullptr, vtbf, Nn, HID, 2);
    k_attn<<<dim3(Nn / 128, HEADS), 256, 0, stream>>>(qbf, kbf, vtbf, beta, i, obf);
    k_gemm<HID><<<dim3(HID / 64, Nn / 16), 32, 0, stream>>>(obf, wfp + o, bf_ + i * HID, tmp, nullptr, Nn, HID, 0);
    k_ln<<<Nn, HID, 0, stream>>>(tmp, h, glob_g, glob_b, i * HID, h, hbf);
  }

  // out = h @ out_w + out_b
  k_gemm<HID><<<dim3(OUTC / 64, Nn / 16), 32, 0, stream>>>(hbf, outp, out_b, (float*)d_out, nullptr, Nn, OUTC, 0);
}
